// MiniGPT_73332271611913
// MI455X (gfx1250) — compile-verified, matrix operand order checked by reading, not run on hardware
//
#include <hip/hip_runtime.h>
#include <hip/hip_bf16.h>

typedef __attribute__((ext_vector_type(16))) _Float16 v16h;
typedef __attribute__((ext_vector_type(8)))  _Float16 h8;
typedef __attribute__((ext_vector_type(4)))  _Float16 h4;
typedef __attribute__((ext_vector_type(8)))  float    v8f;

#define DEV __device__ __forceinline__

DEV v8f wmma32(v16h a, v16h b, v8f c) {
  // D = A(16x32 f16) x B(32x16 f16) + C(16x16 f32)
  return __builtin_amdgcn_wmma_f32_16x16x32_f16(false, a, false, b, (short)0, c, false, false);
}

// Async 16B global -> LDS copy (per active lane), tracked by ASYNCcnt.
DEV void async_copy_b128(unsigned lds_off, const void* gaddr) {
  asm volatile("global_load_async_to_lds_b128 %0, %1, off"
               :: "v"(lds_off), "v"(gaddr) : "memory");
}
DEV void wait_asynccnt0() {
  asm volatile("s_wait_asynccnt 0" ::: "memory");
}

// Load one 16x32 f16 fragment (A-operand layout; B-operand of W[N,K] rows is identical)
// lane l holds row (l&15); k-chunks {kb..kb+7, kb+16..kb+23}, kb=(l>>4)*8
DEV v16h load_frag(const _Float16* base, int ld, int row, int col, int lane) {
  const int r  = row + (lane & 15);
  const int kb = (lane >> 4) << 3;
  const _Float16* p = base + (size_t)r * (size_t)ld + col + kb;
  h8 lo = *(const h8*)(p);
  h8 hi = *(const h8*)(p + 16);
  return __builtin_shufflevector(lo, hi, 0,1,2,3,4,5,6,7,8,9,10,11,12,13,14,15);
}

// ---------------------------------------------------------------------------
// GEMM: C[M,N] = A[M,K](f16) * W[N,K](fp32, converted to f16 in LDS)^T
// mode 0: fp32 store to Cf; mode 1: f16(gelu(acc+bias)) to Ch; mode 2: f16 to Ch
// A tile staged via GLOBAL_LOAD_ASYNC_TO_LDS_B128; W tile via load/convert/ds_store.
// ---------------------------------------------------------------------------
__global__ __launch_bounds__(256) void gemm_f16w32(
    const _Float16* __restrict__ A, const float* __restrict__ W,
    int M, int N, int K,
    float* __restrict__ Cf, _Float16* __restrict__ Ch,
    const float* __restrict__ bias, int mode)
{
  __shared__ _Float16 As[128][32];
  __shared__ _Float16 Bs[128][32];
  const int tid  = threadIdx.x;
  const int lane = tid & 31;
  const int wave = tid >> 5;
  const int wm = wave & 3;       // 4 waves along M (32 rows each)
  const int wn = wave >> 2;      // 2 waves along N (64 cols each)
  const int row0 = blockIdx.y * 128, col0 = blockIdx.x * 128;
  v8f acc[2][4] = {};

  const int ar = tid >> 2;           // 0..63   (A: 4 threads/row, 8 halves each)
  const int ac = (tid & 3) << 3;
  const int br = tid >> 3;           // 0..31   (B: 8 threads/row, 4 floats each)
  const int bc = (tid & 7) << 2;

  const unsigned ldsA0 = (unsigned)(size_t)&As[ar][ac];
  const unsigned ldsA1 = (unsigned)(size_t)&As[ar + 64][ac];

  for (int kk = 0; kk < K; kk += 32) {
    // --- A tile: two async 16B copies per lane, no VGPR staging ---
    async_copy_b128(ldsA0, A + (size_t)(row0 + ar) * K + kk + ac);
    async_copy_b128(ldsA1, A + (size_t)(row0 + ar + 64) * K + kk + ac);
    // --- W tile: gather 4 float4 in flight, then convert + ds_store ---
    float4 f[4];
    #pragma unroll
    for (int it = 0; it < 4; ++it)
      f[it] = *(const float4*)(W + (size_t)(col0 + br + it * 32) * K + kk + bc);
    #pragma unroll
    for (int it = 0; it < 4; ++it) {
      h4 hh;
      hh[0] = (_Float16)f[it].x; hh[1] = (_Float16)f[it].y;
      hh[2] = (_Float16)f[it].z; hh[3] = (_Float16)f[it].w;
      *(h4*)&Bs[br + it * 32][bc] = hh;
    }
    if (kk + 32 < K) {
      __builtin_prefetch(W + (size_t)(col0 + br) * K + kk + 32 + bc, 0, 1);
    }
    wait_asynccnt0();
    __syncthreads();
    v16h af[2], bf[4];
    #pragma unroll
    for (int i = 0; i < 2; ++i) af[i] = load_frag(&As[0][0], 32, wm * 32 + i * 16, 0, lane);
    #pragma unroll
    for (int j = 0; j < 4; ++j) bf[j] = load_frag(&Bs[0][0], 32, wn * 64 + j * 16, 0, lane);
    #pragma unroll
    for (int i = 0; i < 2; ++i)
      #pragma unroll
      for (int j = 0; j < 4; ++j)
        acc[i][j] = wmma32(af[i], bf[j], acc[i][j]);
    __syncthreads();
  }

  const int cl = lane & 15;
  const int rl = (lane >> 4) << 3;
  #pragma unroll
  for (int i = 0; i < 2; ++i)
    #pragma unroll
    for (int j = 0; j < 4; ++j) {
      const int col = col0 + wn * 64 + j * 16 + cl;
      size_t base = (size_t)(row0 + wm * 32 + i * 16 + rl) * (size_t)N + col;
      #pragma unroll
      for (int e = 0; e < 8; ++e) {
        float v = acc[i][j][e];
        size_t o = base + (size_t)e * (size_t)N;
        if (mode == 0) {
          Cf[o] = v;
        } else if (mode == 1) {
          v += bias[col];
          float g = 0.5f * v * (1.0f + erff(v * 0.70710678118654752f));
          Ch[o] = (_Float16)g;
        } else {
          Ch[o] = (_Float16)v;
        }
      }
    }
}

// ---------------------------------------------------------------------------
// Flash attention, one wave per 16-query tile, transposed-S formulation:
//   S^T = K * Q^T  (keys on rows, queries on lanes)  -> per-lane softmax
//   O^T = V^T * P^T (P^T built from S^T accumulators by per-lane f16 packing)
// ---------------------------------------------------------------------------
__global__ __launch_bounds__(256) void attn_flash(
    const _Float16* __restrict__ qkv,   // [B*T, 3*D] f16
    const _Float16* __restrict__ vt,    // [B*H, HD, T] f16 (V transposed)
    _Float16* __restrict__ out)         // [B*T, D] f16
{
  const int T = 2048, D = 1024, HD = 64;
  const int lane = threadIdx.x & 31;
  const int gw = (blockIdx.x * 256 + threadIdx.x) >> 5;
  const int qt = gw & 127;               // T/16 tiles
  const int bh = gw >> 7;
  const int b = bh >> 4, h = bh & 15;
  const int q0 = qt << 4;

  const _Float16* qbase = qkv + (size_t)b * T * 3 * D + (size_t)h * HD;
  const _Float16* kbase = qbase + D;
  const _Float16* vbase = vt + (size_t)bh * HD * T;

  const v16h qf0 = load_frag(qbase, 3 * D, q0, 0, lane);
  const v16h qf1 = load_frag(qbase, 3 * D, q0, 32, lane);

  v8f o[4] = {};
  float m_cur = -1e30f, l_cur = 0.0f;
  const int q = q0 + (lane & 15);
  const int rbase = (lane >> 4) << 3;
  const int ntiles = (q0 + 47) >> 5;     // keys 0..q0+15 in tiles of 32

  for (int kt = 0; kt < ntiles; ++kt) {
    const int k0 = kt << 5;
    v16h ka0 = load_frag(kbase, 3 * D, k0, 0, lane);
    v16h ka1 = load_frag(kbase, 3 * D, k0, 32, lane);
    v16h kc0 = load_frag(kbase, 3 * D, k0 + 16, 0, lane);
    v16h kc1 = load_frag(kbase, 3 * D, k0 + 16, 32, lane);
    v8f s_lo = {}, s_hi = {};
    s_lo = wmma32(ka0, qf0, s_lo);
    s_lo = wmma32(ka1, qf1, s_lo);
    s_hi = wmma32(kc0, qf0, s_hi);
    s_hi = wmma32(kc1, qf1, s_hi);

    float sl[8], sh[8];
    float mt = -1e30f;
    #pragma unroll
    for (int e = 0; e < 8; ++e) {
      const int klo = k0 + rbase + e;
      float a = s_lo[e] * 0.125f;        // HD^-0.5
      float c = s_hi[e] * 0.125f;
      sl[e] = (klo <= q) ? a : -1e30f;
      sh[e] = (klo + 16 <= q) ? c : -1e30f;
      mt = fmaxf(mt, fmaxf(sl[e], sh[e]));
    }
    mt = fmaxf(mt, __shfl_xor(mt, 16));
    const float m_new = fmaxf(m_cur, mt);
    const float alpha = __expf(m_cur - m_new);
    float rs = 0.0f;
    v16h pB;
    #pragma unroll
    for (int e = 0; e < 8; ++e) {
      float pl = __expf(sl[e] - m_new);
      float ph = __expf(sh[e] - m_new);
      rs += pl + ph;
      pB[e]     = (_Float16)pl;          // keys kbase..+7  -> B-operand v0..v3
      pB[e + 8] = (_Float16)ph;          // keys +16..+23   -> B-operand v4..v7
    }
    rs += __shfl_xor(rs, 16);
    l_cur = l_cur * alpha + rs;
    m_cur = m_new;
    #pragma unroll
    for (int c2 = 0; c2 < 4; ++c2)
      #pragma unroll
      for (int e = 0; e < 8; ++e)
        o[c2][e] *= alpha;
    #pragma unroll
    for (int c2 = 0; c2 < 4; ++c2) {
      v16h vf = load_frag(vbase, T, c2 * 16, k0, lane);   // V^T rows = head dims
      o[c2] = wmma32(vf, pB, o[c2]);
    }
  }

  const float inv = 1.0f / l_cur;
  _Float16* orow = out + (size_t)(b * T + q) * D + h * HD + rbase;
  #pragma unroll
  for (int c2 = 0; c2 < 4; ++c2) {
    h8 st;
    #pragma unroll
    for (int e = 0; e < 8; ++e) st[e] = (_Float16)(o[c2][e] * inv);
    *(h8*)(orow + c2 * 16) = st;
  }
}

// ---------------------------------------------------------------------------
// Elementwise / reshape helpers
// ---------------------------------------------------------------------------
__global__ __launch_bounds__(256) void embed_k(
    const int* __restrict__ idx, const float* __restrict__ tok,
    const float* __restrict__ pos, float* __restrict__ x)
{
  const int row = blockIdx.x;            // 0..8191
  const int t = row & 2047;
  const int c = threadIdx.x * 4;
  const int token = idx[row];
  float4 a = *(const float4*)(tok + (size_t)token * 1024 + c);
  float4 p = *(const float4*)(pos + (size_t)t * 1024 + c);
  a.x += p.x; a.y += p.y; a.z += p.z; a.w += p.w;
  *(float4*)(x + (size_t)row * 1024 + c) = a;
}

__global__ __launch_bounds__(256) void layernorm_to_f16(
    const float* __restrict__ x, const float* __restrict__ g,
    const float* __restrict__ bta, _Float16* __restrict__ out)
{
  const int row = blockIdx.x;
  const int tid = threadIdx.x;
  const int lane = tid & 31, wave = tid >> 5;
  float4 v = *(const float4*)(x + (size_t)row * 1024 + tid * 4);
  float s  = v.x + v.y + v.z + v.w;
  float ss = v.x * v.x + v.y * v.y + v.z * v.z + v.w * v.w;
  __shared__ float red[8][2];
  #pragma unroll
  for (int off = 16; off > 0; off >>= 1) {
    s  += __shfl_down(s, off);
    ss += __shfl_down(ss, off);
  }
  if (lane == 0) { red[wave][0] = s; red[wave][1] = ss; }
  __syncthreads();
  float tot = 0.f, tot2 = 0.f;
  #pragma unroll
  for (int w = 0; w < 8; ++w) { tot += red[w][0]; tot2 += red[w][1]; }
  const float mu  = tot * (1.0f / 1024.0f);
  const float var = tot2 * (1.0f / 1024.0f) - mu * mu;
  const float rsg = rsqrtf(var + 1e-5f);
  float4 gv = *(const float4*)(g + tid * 4);
  float4 bv = *(const float4*)(bta + tid * 4);
  h4 o;
  o[0] = (_Float16)((v.x - mu) * rsg * gv.x + bv.x);
  o[1] = (_Float16)((v.y - mu) * rsg * gv.y + bv.y);
  o[2] = (_Float16)((v.z - mu) * rsg * gv.z + bv.z);
  o[3] = (_Float16)((v.w - mu) * rsg * gv.w + bv.w);
  *(h4*)(out + (size_t)row * 1024 + tid * 4) = o;
}

__global__ __launch_bounds__(256) void packvt(
    const _Float16* __restrict__ qkv, _Float16* __restrict__ vt)
{
  // vt[bh][d][t] = qkv[(b*T+t)*3072 + 2048 + h*64 + d]
  size_t i = (size_t)blockIdx.x * 256 + threadIdx.x;   // B*H*64*2048 threads
  const int t = (int)(i & 2047);
  size_t r = i >> 11;
  const int d = (int)(r & 63);
  const size_t bh = r >> 6;
  const int b = (int)(bh >> 4), h = (int)(bh & 15);
  vt[i] = qkv[(size_t)(b * 2048 + t) * 3072 + 2048 + h * 64 + d];
}

__global__ __launch_bounds__(256) void add_res(
    float* __restrict__ x, const float* __restrict__ r, size_t n)
{
  size_t i = (size_t)blockIdx.x * 256 + threadIdx.x;
  const size_t stride = (size_t)gridDim.x * 256;
  for (; i < n; i += stride) x[i] += r[i];
}

__global__ __launch_bounds__(256) void add_res_bias(
    float* __restrict__ x, const float* __restrict__ r,
    const float* __restrict__ bias, size_t n)
{
  size_t i = (size_t)blockIdx.x * 256 + threadIdx.x;
  const size_t stride = (size_t)gridDim.x * 256;
  for (; i < n; i += stride) x[i] += r[i] + bias[i & 1023];
}

// ---------------------------------------------------------------------------
extern "C" void kernel_launch(void* const* d_in, const int* in_sizes, int n_in,
                              void* d_out, int out_size, void* d_ws, size_t ws_size,
                              hipStream_t stream)
{
  (void)in_sizes; (void)n_in; (void)out_size; (void)ws_size;
  const int B = 4, T = 2048, D = 1024, M = B * T;

  const int*   idx    = (const int*)d_in[0];
  const float* tok    = (const float*)d_in[1];
  const float* pos    = (const float*)d_in[2];
  const float* head_w = (const float*)d_in[3];
  const float* lnf_g  = (const float*)d_in[4];
  const float* lnf_b  = (const float*)d_in[5];

  char* ws = (char*)d_ws;
  float*    x      = (float*)   (ws + 0ull);            // 32 MB  [8192,1024] f32
  _Float16* ln_h   = (_Float16*)(ws + 33554432ull);     // 16 MB  [8192,1024] f16
  _Float16* qkv_h  = (_Float16*)(ws + 50331648ull);     // 48 MB  [8192,3072] f16
  _Float16* vt_h   = (_Float16*)(ws + 100663296ull);    // 16 MB  [64,64,2048] f16
  _Float16* attn_h = (_Float16*)(ws + 117440512ull);    // 16 MB  [8192,1024] f16
  _Float16* h4_h   = (_Float16*)(ws + 134217728ull);    // 64 MB  [8192,4096] f16
  float*    gout   = (float*)   (ws + 201326592ull);    // 32 MB  [8192,1024] f32

  embed_k<<<M, 256, 0, stream>>>(idx, tok, pos, x);

  for (int l = 0; l < 2; ++l) {
    const float* ln1_g  = (const float*)d_in[6 + l * 10 + 0];
    const float* ln1_b  = (const float*)d_in[6 + l * 10 + 1];
    const float* qkv_w  = (const float*)d_in[6 + l * 10 + 2];
    const float* proj_w = (const float*)d_in[6 + l * 10 + 3];
    const float* ln2_g  = (const float*)d_in[6 + l * 10 + 4];
    const float* ln2_b  = (const float*)d_in[6 + l * 10 + 5];
    const float* fc1_w  = (const float*)d_in[6 + l * 10 + 6];
    const float* fc1_b  = (const float*)d_in[6 + l * 10 + 7];
    const float* fc2_w  = (const float*)d_in[6 + l * 10 + 8];
    const float* fc2_b  = (const float*)d_in[6 + l * 10 + 9];

    layernorm_to_f16<<<M, 256, 0, stream>>>(x, ln1_g, ln1_b, ln_h);
    gemm_f16w32<<<dim3(3072 / 128, M / 128), 256, 0, stream>>>(
        ln_h, qkv_w, M, 3072, 1024, nullptr, qkv_h, nullptr, 2);
    packvt<<<(B * 16 * 64 * T) / 256, 256, 0, stream>>>(qkv_h, vt_h);
    attn_flash<<<(B * 16 * (T / 16)) / 8, 256, 0, stream>>>(qkv_h, vt_h, attn_h);
    gemm_f16w32<<<dim3(1024 / 128, M / 128), 256, 0, stream>>>(
        attn_h, proj_w, M, 1024, 1024, gout, nullptr, nullptr, 0);
    add_res<<<4096, 256, 0, stream>>>(x, gout, (size_t)M * 1024);
    layernorm_to_f16<<<M, 256, 0, stream>>>(x, ln2_g, ln2_b, ln_h);
    gemm_f16w32<<<dim3(4096 / 128, M / 128), 256, 0, stream>>>(
        ln_h, fc1_w, M, 4096, 1024, nullptr, h4_h, fc1_b, 1);
    gemm_f16w32<<<dim3(1024 / 128, M / 128), 256, 0, stream>>>(
        h4_h, fc2_w, M, 1024, 4096, gout, nullptr, nullptr, 0);
    add_res_bias<<<4096, 256, 0, stream>>>(x, gout, fc2_b, (size_t)M * 1024);
  }

  layernorm_to_f16<<<M, 256, 0, stream>>>(x, lnf_g, lnf_b, ln_h);
  gemm_f16w32<<<dim3(32000 / 128, M / 128), 256, 0, stream>>>(
      ln_h, head_w, M, 32000, 1024, (float*)d_out, nullptr, nullptr, 0);
}
